// NodeModel_input_24773371363899
// MI455X (gfx1250) — matrix operand order repaired and checked by reading, not compiled, not run on hardware
//
#include <hip/hip_runtime.h>

// ---------------------------------------------------------------------------
// CDNA5 (gfx1250, wave32) fused GNN NodeModel:
//   edge:  h = relu(concat(x[row], edge_attr) @ W1a + b1a) @ W1b + b1b
//          atomic scatter-add h into summed[col], count into counts[col]
//   node:  out = relu(concat(x, summed/max(counts,1)) @ W2a + b2a) @ W2b + b2b
// Matmuls via v_wmma_f32_16x16x32_bf16 (bf16 in, f32 accumulate).
// ---------------------------------------------------------------------------

typedef __attribute__((ext_vector_type(16))) __bf16 v16bf;
typedef __attribute__((ext_vector_type(8)))  float  v8f;

struct B32Bytes { uint4 lo, hi; };   // 32 bytes == one v16bf fragment

#define WMMA_BF16(a, b, c) \
  __builtin_amdgcn_wmma_f32_16x16x32_bf16(false, (a), false, (b), (short)0, (c), false, false)

// B fragment (32x16, K x N) from LDS-resident transposed bf16 weights
// wt[n*ldk + k]: lane L -> N = n0 + L%16, K = k0 + (e%8) + 8*(L/16) + 16*(e/8)
// (e = 0..15) => two 16-byte LDS loads.
__device__ __forceinline__ v16bf bfrag_lds(const __bf16* wt, int ldk, int k0,
                                           int n0, int lane) {
  int n = n0 + (lane & 15);
  int k = k0 + ((lane >> 4) << 3);
  const __bf16* p = wt + n * ldk + k;
  B32Bytes raw;
  raw.lo = *reinterpret_cast<const uint4*>(p);
  raw.hi = *reinterpret_cast<const uint4*>(p + 16);
  return __builtin_bit_cast(v16bf, raw);
}

// A fragment (16x32, M x K) from a global f32 row (64 contiguous floats),
// scaled by s, converted to bf16. Per ISA: element e of lane L holds
// K = (e%8) + 8*(L/16) + 16*(e/8)  (row M = L%16 fixed per lane).
__device__ __forceinline__ v16bf afrag_f32(const float* p, int k0, int h,
                                           float s) {
  const float* q = p + k0 + 8 * h;
  float4 f0 = *reinterpret_cast<const float4*>(q);
  float4 f1 = *reinterpret_cast<const float4*>(q + 4);
  float4 f2 = *reinterpret_cast<const float4*>(q + 16);
  float4 f3 = *reinterpret_cast<const float4*>(q + 20);
  v16bf a;
  a[0]  = (__bf16)(f0.x * s); a[1]  = (__bf16)(f0.y * s);
  a[2]  = (__bf16)(f0.z * s); a[3]  = (__bf16)(f0.w * s);
  a[4]  = (__bf16)(f1.x * s); a[5]  = (__bf16)(f1.y * s);
  a[6]  = (__bf16)(f1.z * s); a[7]  = (__bf16)(f1.w * s);
  a[8]  = (__bf16)(f2.x * s); a[9]  = (__bf16)(f2.y * s);
  a[10] = (__bf16)(f2.z * s); a[11] = (__bf16)(f2.w * s);
  a[12] = (__bf16)(f3.x * s); a[13] = (__bf16)(f3.y * s);
  a[14] = (__bf16)(f3.z * s); a[15] = (__bf16)(f3.w * s);
  return a;
}

// A fragment from LDS-staged bf16 activations (row-major [16][64])
__device__ __forceinline__ v16bf afrag_ldsbf(const __bf16* hb, int r, int k0,
                                             int h) {
  const __bf16* q = hb + r * 64 + k0 + 8 * h;
  B32Bytes raw;
  raw.lo = *reinterpret_cast<const uint4*>(q);
  raw.hi = *reinterpret_cast<const uint4*>(q + 16);
  return __builtin_bit_cast(v16bf, raw);
}

__global__ void zero_ws_kernel(float* __restrict__ ws, long long n) {
  long long i = (long long)blockIdx.x * blockDim.x + threadIdx.x;
  if (i < n) ws[i] = 0.0f;
}

// ------------------------- edge MLP + scatter ------------------------------
__global__ __launch_bounds__(256) void edge_mlp_scatter_kernel(
    const float* __restrict__ x, const int* __restrict__ row,
    const int* __restrict__ col, const float* __restrict__ ea,
    const float* __restrict__ w1a, const float* __restrict__ b1a,
    const float* __restrict__ w1b, const float* __restrict__ b1b,
    float* __restrict__ summed, float* __restrict__ counts, int E) {
  __shared__ __attribute__((aligned(16))) __bf16 wtA[64 * 128]; // [n][k] of w1a
  __shared__ __attribute__((aligned(16))) __bf16 wtB[64 * 64];  // [n][k] of w1b
  __shared__ __attribute__((aligned(16))) __bf16 hbuf[8 * 16 * 64];

  int t = threadIdx.x;
  for (int i = t; i < 128 * 64; i += 256) {
    int k = i >> 6, n = i & 63;
    wtA[n * 128 + k] = (__bf16)w1a[i];
  }
  for (int i = t; i < 64 * 64; i += 256) {
    int k = i >> 6, n = i & 63;
    wtB[n * 64 + k] = (__bf16)w1b[i];
  }
  __syncthreads();

  int wave = t >> 5, lane = t & 31;
  int tile = blockIdx.x * 8 + wave;
  int base = tile * 16;
  if (base >= E) return;
  int r = lane & 15, h = lane >> 4;
  int e = base + r;
  if (e >= E) e = E - 1;

  const float* xrow = x + (long long)row[e] * 64;
  const float* erow = ea + (long long)e * 64;

  // A fragments: K 0..63 from x[row], K 64..127 from edge_attr
  v16bf afr[4];
  afr[0] = afrag_f32(xrow, 0, h, 1.0f);
  afr[1] = afrag_f32(xrow, 32, h, 1.0f);
  afr[2] = afrag_f32(erow, 0, h, 1.0f);
  afr[3] = afrag_f32(erow, 32, h, 1.0f);

  // layer 1: [16x128] @ [128x64] + b1a
  v8f acc[4];
#pragma unroll
  for (int nb = 0; nb < 4; ++nb) {
    float bv = b1a[nb * 16 + r];
#pragma unroll
    for (int j = 0; j < 8; ++j) acc[nb][j] = bv;
  }
#pragma unroll
  for (int kb = 0; kb < 4; ++kb)
#pragma unroll
    for (int nb = 0; nb < 4; ++nb)
      acc[nb] = WMMA_BF16(afr[kb], bfrag_lds(wtA, 128, kb * 32, nb * 16, lane),
                          acc[nb]);

  // ReLU, stage to LDS as bf16 (lane transpose D-layout -> A-layout)
  __bf16* hb = hbuf + wave * (16 * 64);
#pragma unroll
  for (int nb = 0; nb < 4; ++nb)
#pragma unroll
    for (int j = 0; j < 8; ++j) {
      float v = acc[nb][j];
      v = v > 0.0f ? v : 0.0f;
      hb[(j + 8 * h) * 64 + nb * 16 + r] = (__bf16)v;
    }
  asm volatile("s_wait_dscnt 0" ::: "memory"); // same-wave cross-lane via LDS

  // layer 2: [16x64] @ [64x64] + b1b
  v8f acc2[4];
#pragma unroll
  for (int nb = 0; nb < 4; ++nb) {
    float bv = b1b[nb * 16 + r];
#pragma unroll
    for (int j = 0; j < 8; ++j) acc2[nb][j] = bv;
  }
#pragma unroll
  for (int kb = 0; kb < 2; ++kb) {
    v16bf a2 = afrag_ldsbf(hb, r, kb * 32, h);
#pragma unroll
    for (int nb = 0; nb < 4; ++nb)
      acc2[nb] = WMMA_BF16(a2, bfrag_lds(wtB, 64, kb * 32, nb * 16, lane),
                           acc2[nb]);
  }

  // atomic scatter-add into summed[col] (L2-resident accumulator)
  int dst[8];
  bool valid[8];
#pragma unroll
  for (int j = 0; j < 8; ++j) {
    int ee = base + j + 8 * h;
    valid[j] = ee < E;
    dst[j] = col[valid[j] ? ee : (E - 1)];
  }
#pragma unroll
  for (int nb = 0; nb < 4; ++nb)
#pragma unroll
    for (int j = 0; j < 8; ++j)
      if (valid[j])
        atomicAdd(&summed[(long long)dst[j] * 64 + nb * 16 + r], acc2[nb][j]);

  if (h == 0 && (base + r) < E) atomicAdd(&counts[col[base + r]], 1.0f);
}

// ------------------------------ node MLP -----------------------------------
__global__ __launch_bounds__(256) void node_mlp_kernel(
    const float* __restrict__ x, const float* __restrict__ summed,
    const float* __restrict__ counts, const float* __restrict__ w2a,
    const float* __restrict__ b2a, const float* __restrict__ w2b,
    const float* __restrict__ b2b, float* __restrict__ out, int N) {
  __shared__ __attribute__((aligned(16))) __bf16 wtA[64 * 128];
  __shared__ __attribute__((aligned(16))) __bf16 wtB[64 * 64];
  __shared__ __attribute__((aligned(16))) __bf16 hbuf[8 * 16 * 64];

  int t = threadIdx.x;
  for (int i = t; i < 128 * 64; i += 256) {
    int k = i >> 6, n = i & 63;
    wtA[n * 128 + k] = (__bf16)w2a[i];
  }
  for (int i = t; i < 64 * 64; i += 256) {
    int k = i >> 6, n = i & 63;
    wtB[n * 64 + k] = (__bf16)w2b[i];
  }
  __syncthreads();

  int wave = t >> 5, lane = t & 31;
  int tile = blockIdx.x * 8 + wave;
  int base = tile * 16;
  if (base >= N) return;
  int r = lane & 15, h = lane >> 4;
  int node = base + r;
  if (node >= N) node = N - 1;

  const float* xrow = x + (long long)node * 64;
  const float* srow = summed + (long long)node * 64;
  float c = counts[node];
  float inv = 1.0f / (c > 1.0f ? c : 1.0f); // scatter_mean: /max(count,1)

  v16bf afr[4];
  afr[0] = afrag_f32(xrow, 0, h, 1.0f);
  afr[1] = afrag_f32(xrow, 32, h, 1.0f);
  afr[2] = afrag_f32(srow, 0, h, inv);
  afr[3] = afrag_f32(srow, 32, h, inv);

  v8f acc[4];
#pragma unroll
  for (int nb = 0; nb < 4; ++nb) {
    float bv = b2a[nb * 16 + r];
#pragma unroll
    for (int j = 0; j < 8; ++j) acc[nb][j] = bv;
  }
#pragma unroll
  for (int kb = 0; kb < 4; ++kb)
#pragma unroll
    for (int nb = 0; nb < 4; ++nb)
      acc[nb] = WMMA_BF16(afr[kb], bfrag_lds(wtA, 128, kb * 32, nb * 16, lane),
                          acc[nb]);

  __bf16* hb = hbuf + wave * (16 * 64);
#pragma unroll
  for (int nb = 0; nb < 4; ++nb)
#pragma unroll
    for (int j = 0; j < 8; ++j) {
      float v = acc[nb][j];
      v = v > 0.0f ? v : 0.0f;
      hb[(j + 8 * h) * 64 + nb * 16 + r] = (__bf16)v;
    }
  asm volatile("s_wait_dscnt 0" ::: "memory");

  v8f acc2[4];
#pragma unroll
  for (int nb = 0; nb < 4; ++nb) {
    float bv = b2b[nb * 16 + r];
#pragma unroll
    for (int j = 0; j < 8; ++j) acc2[nb][j] = bv;
  }
#pragma unroll
  for (int kb = 0; kb < 2; ++kb) {
    v16bf a2 = afrag_ldsbf(hb, r, kb * 32, h);
#pragma unroll
    for (int nb = 0; nb < 4; ++nb)
      acc2[nb] = WMMA_BF16(a2, bfrag_lds(wtB, 64, kb * 32, nb * 16, lane),
                           acc2[nb]);
  }

#pragma unroll
  for (int nb = 0; nb < 4; ++nb)
#pragma unroll
    for (int j = 0; j < 8; ++j) {
      int m = base + j + 8 * h;
      if (m < N) out[(long long)m * 64 + nb * 16 + r] = acc2[nb][j];
    }
}

// ------------------------------ launcher -----------------------------------
extern "C" void kernel_launch(void* const* d_in, const int* in_sizes, int n_in,
                              void* d_out, int out_size, void* d_ws,
                              size_t ws_size, hipStream_t stream) {
  const float* x   = (const float*)d_in[0];
  const int*   ei  = (const int*)d_in[1];
  const float* ea  = (const float*)d_in[2];
  // d_in[3] = u, d_in[4] = batch : unused by NodeModel reference
  const float* w1a = (const float*)d_in[5];
  const float* b1a = (const float*)d_in[6];
  const float* w1b = (const float*)d_in[7];
  const float* b1b = (const float*)d_in[8];
  const float* w2a = (const float*)d_in[9];
  const float* b2a = (const float*)d_in[10];
  const float* w2b = (const float*)d_in[11];
  const float* b2b = (const float*)d_in[12];

  int N = in_sizes[0] / 64;
  int E = in_sizes[1] / 2;
  const int* row = ei;
  const int* col = ei + E;

  float* summed = (float*)d_ws;                 // [N][64]
  float* counts = summed + (long long)N * 64;   // [N]

  long long zn = (long long)N * 64 + N;
  zero_ws_kernel<<<(int)((zn + 255) / 256), 256, 0, stream>>>(summed, zn);

  int etiles = (E + 15) / 16;
  edge_mlp_scatter_kernel<<<(etiles + 7) / 8, 256, 0, stream>>>(
      x, row, col, ea, w1a, b1a, w1b, b1b, summed, counts, E);

  int ntiles = (N + 15) / 16;
  node_mlp_kernel<<<(ntiles + 7) / 8, 256, 0, stream>>>(
      x, summed, counts, w2a, b2a, w2b, b2b, (float*)d_out, N);
}
